// RWKV7Block_35167192219992
// MI455X (gfx1250) — matrix operand order repaired and checked by reading, not compiled, not run on hardware
//
#include <hip/hip_runtime.h>

typedef __attribute__((ext_vector_type(16))) __bf16        v16bf;
typedef __attribute__((ext_vector_type(8)))  float         v8f;
typedef __attribute__((ext_vector_type(4)))  unsigned int  v4u;

#define TILE_M 128
#define TILE_N 128
#define TILE_K 32
#define LDSS   40   // padded LDS row stride in bf16 elements (80B, 16B-aligned)

// ---------------------------------------------------------------------------
// f32 -> bf16 conversion (weights)
// ---------------------------------------------------------------------------
__global__ __launch_bounds__(256)
void cvt_bf16_kernel(const float* __restrict__ in, __bf16* __restrict__ out, int n)
{
    int i = blockIdx.x * 256 + threadIdx.x;
#pragma unroll
    for (int j = 0; j < 4; ++j) {
        int idx = i * 4 + j;
        if (idx < n) out[idx] = (__bf16)in[idx];
    }
}

// ---------------------------------------------------------------------------
// Fused LayerNorm + token-shift mix. One 256-thread block per row.
// Recomputes LN stats for the previous row (token shift) instead of
// materializing h. NOUT==3 -> r/k/v mixes; NOUT==1 -> channel-mix.
// ---------------------------------------------------------------------------
template <int NOUT>
__global__ __launch_bounds__(256)
void ln_mix_kernel(const float* __restrict__ x,
                   const float* __restrict__ w, const float* __restrict__ bb,
                   const float* __restrict__ m0, const float* __restrict__ m1,
                   const float* __restrict__ m2,
                   __bf16* __restrict__ o0, __bf16* __restrict__ o1,
                   __bf16* __restrict__ o2,
                   int S, int D)
{
    const int row = blockIdx.x;
    const int s   = row % S;
    const bool hasPrev = (s != 0);
    const float* xr = x + (size_t)row * D;
    const float* xp = xr - D;

    float s0 = 0.f, q0 = 0.f, s1 = 0.f, q1 = 0.f;
    for (int j = threadIdx.x; j < D; j += 256) {
        float a = xr[j]; s0 += a; q0 += a * a;
        if (hasPrev) { float p = xp[j]; s1 += p; q1 += p * p; }
    }

    __shared__ float red[4][256];
    const int tid = threadIdx.x;
    red[0][tid] = s0; red[1][tid] = q0; red[2][tid] = s1; red[3][tid] = q1;
    __syncthreads();
    for (int off = 128; off > 0; off >>= 1) {
        if (tid < off) {
#pragma unroll
            for (int c = 0; c < 4; ++c) red[c][tid] += red[c][tid + off];
        }
        __syncthreads();
    }
    const float invD  = 1.0f / (float)D;
    const float mean  = red[0][0] * invD;
    const float var   = red[1][0] * invD - mean * mean;
    const float rstd  = rsqrtf(var + 1e-5f);
    const float meanp = red[2][0] * invD;
    const float varp  = red[3][0] * invD - meanp * meanp;
    const float rstdp = rsqrtf(varp + 1e-5f);

    const size_t rbase = (size_t)row * D;
    for (int j = threadIdx.x; j < D; j += 256) {
        float h = (xr[j] - mean) * rstd * w[j] + bb[j];
        float p = hasPrev ? ((xp[j] - meanp) * rstdp * w[j] + bb[j]) : 0.f;
        float mm0 = m0[j];
        o0[rbase + j] = (__bf16)(h * mm0 + p * (1.f - mm0));
        if constexpr (NOUT == 3) {
            float mm1 = m1[j];
            float mm2 = m2[j];
            o1[rbase + j] = (__bf16)(h * mm1 + p * (1.f - mm1));
            o2[rbase + j] = (__bf16)(h * mm2 + p * (1.f - mm2));
        }
    }
}

// ---------------------------------------------------------------------------
// WKV elementwise linear recurrence. One thread per (b, d) channel,
// coalesced across d; sequential over S.
// out = sigmoid(r) * state  (bonus term folded into t == 0), stored bf16
// (A-operand of the Wo GEMM).
// ---------------------------------------------------------------------------
__global__ __launch_bounds__(256)
void wkv_kernel(const float* __restrict__ r, const float* __restrict__ k,
                const float* __restrict__ v, const float* __restrict__ td,
                const float* __restrict__ tf, __bf16* __restrict__ out,
                int S, int D)
{
    const int gid = blockIdx.x * 256 + threadIdx.x;   // b*D + d
    const int d   = gid % D;
    const float w = __expf(-__expf(td[d]));
    const float u = __expf(tf[d]);

    size_t idx = (size_t)(gid / D) * (size_t)S * D + d;
    // t == 0: state = kv0; out0 = sig(r0) * (state + u*kv0)
    float kv    = k[idx] * v[idx];
    float state = kv;
    float rs    = 1.f / (1.f + __expf(-r[idx]));
    out[idx]    = (__bf16)(rs * (state + u * kv));
    idx += D;
    for (int t = 1; t < S; ++t, idx += D) {
        kv    = k[idx] * v[idx];
        state = state * w + kv;
        rs    = 1.f / (1.f + __expf(-r[idx]));
        out[idx] = (__bf16)(rs * state);
    }
}

// ---------------------------------------------------------------------------
// bf16 WMMA GEMM: C[M,N] = A[M,K] @ B[K,N], f32 accumulate.
// 128x128 tile / block (256 threads = 8 waves, 4(M) x 2(N) wave grid),
// K-step 32, v_wmma_f32_16x16x32_bf16 inner loop (8 WMMAs / wave / step).
// Epilogues: 0 f32 store | 1 out=add+c | 2 sigmoid(c) | 3 bf16 (c*c)
//            4 out = add + rmul*c
// (reference relu(square(x)) == square(x) since square >= 0)
// ---------------------------------------------------------------------------
template <int EPI>
__global__ __launch_bounds__(256)
void gemm_wmma_kernel(const __bf16* __restrict__ A, const __bf16* __restrict__ B,
                      float* __restrict__ outF, __bf16* __restrict__ outB,
                      const float* __restrict__ add, const float* __restrict__ rmul,
                      int M, int N, int K)
{
    __shared__ __align__(16) __bf16 As[TILE_M * LDSS];
    __shared__ __align__(16) __bf16 Bs[TILE_N * LDSS];   // transposed: [n][k]

    const int tid   = threadIdx.x;
    const int wave  = tid >> 5;
    const int lane  = tid & 31;
    const int ll    = lane & 15;
    const int hi    = lane >> 4;
    const int waveM = wave & 3;    // 4 waves along M, 32 rows each
    const int waveN = wave >> 2;   // 2 waves along N, 64 cols each

    const int mBase = blockIdx.y * TILE_M;
    const int nBase = blockIdx.x * TILE_N;

    v8f acc[2][4];
#pragma unroll
    for (int i = 0; i < 2; ++i)
#pragma unroll
        for (int j = 0; j < 4; ++j) {
            v8f z = {0.f, 0.f, 0.f, 0.f, 0.f, 0.f, 0.f, 0.f};
            acc[i][j] = z;
        }

    union Frag { v16bf v; v4u q[2]; };

    for (int k0 = 0; k0 < K; k0 += TILE_K) {
        __syncthreads();
        // --- A tile 128x32 (row-major, b128 global loads -> b128 LDS stores)
#pragma unroll
        for (int it = 0; it < 2; ++it) {
            int e   = (it * 256 + tid) * 8;
            int row = e >> 5;       // /32
            int col = e & 31;
            v4u val = *(const v4u*)(A + (size_t)(mBase + row) * K + k0 + col);
            *(v4u*)(&As[row * LDSS + col]) = val;
        }
        // --- B tile 32x128 (coalesced b128 global loads, transposed scatter)
#pragma unroll
        for (int it = 0; it < 2; ++it) {
            int p  = it * 256 + tid;
            int kk = p >> 4;           // 0..31
            int n8 = (p & 15) * 8;     // 0..120
            v4u bv = *(const v4u*)(B + (size_t)(k0 + kk) * N + nBase + n8);
            union { v4u q; __bf16 h[8]; } u; u.q = bv;
#pragma unroll
            for (int j = 0; j < 8; ++j)
                Bs[(n8 + j) * LDSS + kk] = u.h[j];
        }
        __syncthreads();

        // --- fragments (ISA 16-bit layouts)
        // A (16x32): lanes 0-15: elems 0-7 = K0-7, 8-15 = K16-23; lanes 16-31: +8
        Frag fa[2], fb[4];
#pragma unroll
        for (int i = 0; i < 2; ++i) {
            const char* p = (const char*)As +
                (size_t)((waveM * 32 + i * 16 + ll) * LDSS + hi * 8) * 2;
            fa[i].q[0] = *(const v4u*)p;
            fa[i].q[1] = *(const v4u*)(p + 32);
        }
        // B (32x16): lanes 0-15: elems 0-7 = K0-7, 8-15 = K16-23 of col N=ll; lanes 16-31: +8
#pragma unroll
        for (int j = 0; j < 4; ++j) {
            const char* p = (const char*)Bs +
                (size_t)((waveN * 64 + j * 16 + ll) * LDSS + hi * 8) * 2;
            fb[j].q[0] = *(const v4u*)p;
            fb[j].q[1] = *(const v4u*)(p + 32);
        }
#pragma unroll
        for (int i = 0; i < 2; ++i)
#pragma unroll
            for (int j = 0; j < 4; ++j)
                acc[i][j] = __builtin_amdgcn_wmma_f32_16x16x32_bf16(
                    false, fa[i].v, false, fb[j].v,
                    (short)0, acc[i][j], false, false);
    }

    // --- epilogue (C layout: lane ll = N, VGPR r -> M = r + hi*8)
#pragma unroll
    for (int i = 0; i < 2; ++i) {
#pragma unroll
        for (int j = 0; j < 4; ++j) {
            const int col  = nBase + waveN * 64 + j * 16 + ll;
            const int row0 = mBase + waveM * 32 + i * 16 + hi * 8;
#pragma unroll
            for (int rr = 0; rr < 8; ++rr) {
                const size_t idx = (size_t)(row0 + rr) * N + col;
                const float  c   = acc[i][j][rr];
                if (EPI == 0)      outF[idx] = c;
                else if (EPI == 1) outF[idx] = add[idx] + c;
                else if (EPI == 2) outF[idx] = 1.f / (1.f + __expf(-c));
                else if (EPI == 3) outB[idx] = (__bf16)(c * c);
                else               outF[idx] = add[idx] + rmul[idx] * c;
            }
        }
    }
}

// ---------------------------------------------------------------------------
// Host launch: orchestrates the full RWKV block.
// ---------------------------------------------------------------------------
extern "C" void kernel_launch(void* const* d_in, const int* in_sizes, int n_in,
                              void* d_out, int out_size, void* d_ws, size_t ws_size,
                              hipStream_t stream)
{
    const float* x   = (const float*)d_in[0];
    const float* n1w = (const float*)d_in[1];
    const float* n1b = (const float*)d_in[2];
    const float* n2w = (const float*)d_in[3];
    const float* n2b = (const float*)d_in[4];
    const float* td  = (const float*)d_in[5];
    const float* tf  = (const float*)d_in[6];
    const float* Wr  = (const float*)d_in[7];
    const float* Wk  = (const float*)d_in[8];
    const float* Wv  = (const float*)d_in[9];
    const float* Wo  = (const float*)d_in[10];
    const float* tmr = (const float*)d_in[11];
    const float* tmk = (const float*)d_in[12];
    const float* tmv = (const float*)d_in[13];
    const float* Wcr = (const float*)d_in[14];
    const float* Wck = (const float*)d_in[15];
    const float* Wcv = (const float*)d_in[16];
    const float* cm  = (const float*)d_in[17];

    constexpr int    Bb  = 4, S = 4096, D = 2048;
    constexpr int    ROWS = Bb * S;                  // 16384
    constexpr size_t WDD  = (size_t)D * D;           // 4 Mi elements
    constexpr size_t MK   = (size_t)ROWS * D;        // 32 Mi elements

    // Workspace layout (bf16 region first, then f32 region), with reuse:
    __bf16* wsB  = (__bf16*)d_ws;
    __bf16* w_r  = wsB + 0 * WDD;
    __bf16* w_k  = wsB + 1 * WDD;
    __bf16* w_v  = wsB + 2 * WDD;
    __bf16* w_o  = wsB + 3 * WDD;
    __bf16* w_cr = wsB + 4 * WDD;
    __bf16* w_ck = wsB + 5 * WDD;   // 3*WDD
    __bf16* w_cv = wsB + 8 * WDD;   // 3*WDD
    __bf16* a0   = wsB + 11 * WDD;  // mixed_r -> wkv(bf16)
    __bf16* a1   = a0 + MK;         // mixed_k -> mixed_cm
    __bf16* a2   = a0 + 2 * MK;     // mixed_v
    __bf16* kcm  = a0 + 3 * MK;     // k_cm bf16 [ROWS x 3D]
    float*  fbase = (float*)(a0 + 6 * MK);
    float*  r_f  = fbase;           // r      -> x2
    float*  k_f  = fbase + MK;      // k      -> r_cm
    float*  v_f  = fbase + 2 * MK;  // v
    float*  x2   = r_f;
    float*  rcm  = k_f;
    float*  outF = (float*)d_out;

    dim3 blk(256);

    auto cvt = [&](const float* src, __bf16* dst, size_t n) {
        cvt_bf16_kernel<<<dim3((unsigned)((n + 1023) / 1024)), blk, 0, stream>>>(
            src, dst, (int)n);
    };
    cvt(Wr,  w_r,  WDD);
    cvt(Wk,  w_k,  WDD);
    cvt(Wv,  w_v,  WDD);
    cvt(Wo,  w_o,  WDD);
    cvt(Wcr, w_cr, WDD);
    cvt(Wck, w_ck, 3 * WDD);
    cvt(Wcv, w_cv, 3 * WDD);

    // LN1 + token-shift mixes (r/k/v) -> bf16 A operands
    ln_mix_kernel<3><<<dim3(ROWS), blk, 0, stream>>>(
        x, n1w, n1b, tmr, tmk, tmv, a0, a1, a2, S, D);

    dim3 gD(D / 128, ROWS / 128);      // 16 x 128
    gemm_wmma_kernel<0><<<gD, blk, 0, stream>>>(a0, w_r, r_f, nullptr, nullptr, nullptr, ROWS, D, D);
    gemm_wmma_kernel<0><<<gD, blk, 0, stream>>>(a1, w_k, k_f, nullptr, nullptr, nullptr, ROWS, D, D);
    gemm_wmma_kernel<0><<<gD, blk, 0, stream>>>(a2, w_v, v_f, nullptr, nullptr, nullptr, ROWS, D, D);

    // WKV recurrence -> bf16 (a0 reusable: its GEMM has completed)
    wkv_kernel<<<dim3((Bb * D) / 256), blk, 0, stream>>>(r_f, k_f, v_f, td, tf, a0, S, D);

    // x2 = x + wkv @ Wo   (residual epilogue; r_f dead -> reused as x2)
    gemm_wmma_kernel<1><<<gD, blk, 0, stream>>>(a0, w_o, x2, nullptr, x, nullptr, ROWS, D, D);

    // LN2 + channel-mix shift -> mixed_cm (a1 reusable)
    ln_mix_kernel<1><<<dim3(ROWS), blk, 0, stream>>>(
        x2, n2w, n2b, cm, nullptr, nullptr, a1, nullptr, nullptr, S, D);

    // r_cm = sigmoid(mixed @ Wcr)   (k_f dead -> reused as rcm)
    gemm_wmma_kernel<2><<<gD, blk, 0, stream>>>(a1, w_cr, rcm, nullptr, nullptr, nullptr, ROWS, D, D);

    // k_cm = square(mixed @ Wck)  (relu of a square is identity), bf16
    dim3 g3((3 * D) / 128, ROWS / 128);
    gemm_wmma_kernel<3><<<g3, blk, 0, stream>>>(a1, w_ck, nullptr, kcm, nullptr, nullptr, ROWS, 3 * D, D);

    // out = x2 + r_cm * (k_cm @ Wcv)
    gemm_wmma_kernel<4><<<gD, blk, 0, stream>>>(kcm, w_cv, outF, nullptr, x2, rcm, ROWS, D, 3 * D);
}
